// hierarchical_Feature_spatial_attention_6614249636391
// MI455X (gfx1250) — matrix operand order repaired and checked by reading, not compiled, not run on hardware
//
#include <hip/hip_runtime.h>

// ---------------------------------------------------------------------------
// Fused "spatial attention" = flash-attention over the m axis.
//   scores[m,n] = sum_c yi[c,m]*pi[c,n];  w = softmax_m(scores);
//   out[c,n]    = sum_m yi[c,m]*w[m,n]
// B=2, C=64, M=N=96*96=9216. Online softmax over m-tiles of 64, bf16 WMMA
// (16x16x32) with fp32 accumulation, per-workgroup 64-column n tile.
// ---------------------------------------------------------------------------

typedef __bf16 bf16_t;
typedef bf16_t v16bf __attribute__((ext_vector_type(16)));
typedef float  v8f   __attribute__((ext_vector_type(8)));
typedef int    v8i   __attribute__((ext_vector_type(8)));

#define MDIM  9216
#define CDIM  64
#define NT    64
#define MT    64
#define LDT   66          // padded bf16 row stride: 132B = 33 dwords -> conflict-free
#define OLD   65          // padded fp32 row stride for output buffer
#define LOG2E 1.44269504088896340736f

union BfOp { v8i i; v16bf v; };

__device__ __forceinline__ unsigned short f2bf(float x) {
  unsigned u = __float_as_uint(x);
  u += 0x7fffu + ((u >> 16) & 1u);          // round-to-nearest-even
  return (unsigned short)(u >> 16);
}
__device__ __forceinline__ int pk2bf(float lo, float hi) {
  unsigned a = __float_as_uint(lo); a += 0x7fffu + ((a >> 16) & 1u);
  unsigned b = __float_as_uint(hi); b += 0x7fffu + ((b >> 16) & 1u);
  return (int)((a >> 16) | (b & 0xffff0000u));
}
// exchange with lane^16 (wave32): ds_swizzle SWAPX16 (xor=0x10, and=0x1f)
__device__ __forceinline__ int xor16(int v) {
  return __builtin_amdgcn_ds_swizzle(v, 0x401f);
}
__device__ __forceinline__ float xor16f(float v) {
  return __int_as_float(xor16(__float_as_int(v)));
}

__global__ __launch_bounds__(256)
void hfsa_flash_attn_kernel(const float* __restrict__ yi,
                            const float* __restrict__ pi,
                            float* __restrict__ out) {
  __shared__ unsigned short sYT[MT * LDT];   // yi^T tile: [m_local][c]   (GEMM1 A)
  __shared__ unsigned short sY [CDIM * LDT]; // yi   tile: [c][m_local]   (GEMM2 A)
  __shared__ unsigned short sPT[NT * LDT];   // pi^T tile: [n_local][c]   (GEMM1 B)
  __shared__ float sObuf[CDIM * OLD];        // fused output accumulator (combine)
  __shared__ float sMaxRun[NT], sSumRun[NT], sMaxNew[NT], sScale[NT];
  __shared__ float sMaxPart[2 * NT], sSumPart[2 * NT];

  const int b      = blockIdx.y;
  const int n0     = blockIdx.x * NT;
  const int tid    = threadIdx.x;
  const int lane   = tid & 31;
  const int wave   = tid >> 5;
  const int lane16 = lane & 15;
  const int lhalf  = lane >> 4;      // 0: lanes 0-15, 1: lanes 16-31
  const int nt     = wave & 3;       // which 16-column n sub-tile
  const int mp     = wave >> 2;      // which 32-row m slice within the m-step

  const float* Yb = yi + (size_t)b * CDIM * MDIM;
  const float* Pb = pi + (size_t)b * CDIM * MDIM;

  // ---- stage pi^T tile (once), init softmax running stats -----------------
  {
    const int nl = tid & 63, cc = tid >> 6;
    for (int c = cc; c < CDIM; c += 4)
      sPT[nl * LDT + c] = f2bf(Pb[(size_t)c * MDIM + n0 + nl]);
    if (tid < NT) { sMaxRun[tid] = -3.0e38f; sSumRun[tid] = 0.0f; }
  }

  v8f Oacc[4];                       // 4 c-subtiles x (16x16 fp32), n sub-tile nt
#pragma unroll
  for (int ct = 0; ct < 4; ++ct) Oacc[ct] = (v8f){};

  for (int step = 0; step < MDIM / MT; ++step) {
    const int m0 = step * MT;
    __syncthreads();                 // prev GEMM2 / sSumPart writes complete

    // fold previous step's partial sums into the running denominator
    if (step > 0 && tid < NT)
      sSumRun[tid] = sSumRun[tid] * sScale[tid] + sSumPart[tid] + sSumPart[NT + tid];

    // ---- stage yi m-tile in both orientations (fp32 -> bf16) --------------
    {
      const int ml = tid & 63, cc = tid >> 6;
      for (int c = cc; c < CDIM; c += 4) {
        float v = Yb[(size_t)c * MDIM + m0 + ml];
        unsigned short h = f2bf(v);
        sYT[ml * LDT + c] = h;
        sY [c * LDT + ml] = h;
        if (m0 + MT < MDIM)          // prefetch next m-tile into cache
          __builtin_prefetch(&Yb[(size_t)c * MDIM + m0 + MT + ml], 0, 0);
      }
    }
    __syncthreads();

    // ---- GEMM1: S(32m x 16n) = yi^T x pi, K = C = 64 ----------------------
    v8f S0 = (v8f){}, S1 = (v8f){};
#pragma unroll
    for (int ks = 0; ks < 2; ++ks) {
      BfOp Bop, A0, A1;
      const int prow = 16 * nt + lane16;
      const int cbB  = ks * 32 + lhalf * 16;          // B: lanes<16 k=0..15, >=16 k=16..31
#pragma unroll
      for (int g = 0; g < 8; ++g)
        Bop.i[g] = *(const int*)&sPT[prow * LDT + cbB + 2 * g];
      const int r0 = 32 * mp + lane16;
#pragma unroll
      for (int g = 0; g < 8; ++g) {                   // A: lanes<16 k=0..15, >=16 k=16..31
        const int col = ks * 32 + lhalf * 16 + 2 * g;
        A0.i[g] = *(const int*)&sYT[r0 * LDT + col];
        A1.i[g] = *(const int*)&sYT[(r0 + 16) * LDT + col];
      }
      S0 = __builtin_amdgcn_wmma_f32_16x16x32_bf16(false, A0.v, false, Bop.v,
                                                   (short)0, S0, false, false);
      S1 = __builtin_amdgcn_wmma_f32_16x16x32_bf16(false, A1.v, false, Bop.v,
                                                   (short)0, S1, false, false);
    }

    // ---- per-column max over this wave's 32 m rows ------------------------
    float mx = S0[0];
#pragma unroll
    for (int g = 1; g < 8; ++g) mx = fmaxf(mx, S0[g]);
#pragma unroll
    for (int g = 0; g < 8; ++g) mx = fmaxf(mx, S1[g]);
    mx = fmaxf(mx, xor16f(mx));                       // lane<->lane^16 share column n
    if (lhalf == 0) sMaxPart[mp * NT + 16 * nt + lane16] = mx;
    __syncthreads();

    if (tid < NT) {                                   // combine running max, get scale
      float mo = sMaxRun[tid];
      float mn = fmaxf(mo, fmaxf(sMaxPart[tid], sMaxPart[NT + tid]));
      sMaxNew[tid] = mn;
      sScale[tid]  = exp2f((mo - mn) * LOG2E);
      sMaxRun[tid] = mn;
    }
    __syncthreads();

    const int   ncol = 16 * nt + lane16;
    const float mn   = sMaxNew[ncol];
    const float sc   = sScale[ncol];

    // ---- P = exp(S - max); column sums; pack 32x16 bf16 B operand ---------
    float p0[8], p1[8], ps = 0.0f;
#pragma unroll
    for (int g = 0; g < 8; ++g) {
      p0[g] = exp2f((S0[g] - mn) * LOG2E);
      p1[g] = exp2f((S1[g] - mn) * LOG2E);
      ps += p0[g] + p1[g];
    }
    ps += xor16f(ps);
    if (lhalf == 0) sSumPart[mp * NT + ncol] = ps;

    BfOp Bp;                                          // D-layout -> B-layout via xor16
#pragma unroll
    for (int j = 0; j < 4; ++j) {
      int a  = pk2bf(p0[2 * j], p0[2 * j + 1]);       // own m rows (k 0-7 / 24-31)
      int c  = pk2bf(p1[2 * j], p1[2 * j + 1]);
      int ax = xor16(a), cx = xor16(c);               // partner's m rows
      Bp.i[j]     = (lhalf == 0) ? a  : cx;           // k = 0..7 (lanes<16) / 16..23 (lanes>=16)
      Bp.i[4 + j] = (lhalf == 0) ? ax : c;            // k = 8..15 (lanes<16) / 24..31 (lanes>=16)
    }

    // rescale running output partials (scale depends only on lane's column n)
#pragma unroll
    for (int ct = 0; ct < 4; ++ct)
#pragma unroll
      for (int g = 0; g < 8; ++g) Oacc[ct][g] *= sc;

    // ---- GEMM2 partial: O(16c x 16n) += yi(c, m-slice) x P, K = 32 --------
#pragma unroll
    for (int ct = 0; ct < 4; ++ct) {
      BfOp A;
      const int row = 16 * ct + lane16;
#pragma unroll
      for (int g = 0; g < 8; ++g) {
        const int col = 32 * mp + lhalf * 16 + 2 * g;
        A.i[g] = *(const int*)&sY[row * LDT + col];
      }
      Oacc[ct] = __builtin_amdgcn_wmma_f32_16x16x32_bf16(false, A.v, false, Bp.v,
                                                         (short)0, Oacc[ct], false, false);
    }
  }

  // ---- final denominator fold, cross-wave (mp0 + mp1) combine, writeout ----
  __syncthreads();
  if (tid < NT)
    sSumRun[tid] = sSumRun[tid] * sScale[tid] + sSumPart[tid] + sSumPart[NT + tid];
  if (mp == 0) {
#pragma unroll
    for (int ct = 0; ct < 4; ++ct)
#pragma unroll
      for (int g = 0; g < 8; ++g)
        sObuf[(16 * ct + lhalf * 8 + g) * OLD + 16 * nt + lane16] = Oacc[ct][g];
  }
  __syncthreads();
  if (mp == 1) {
#pragma unroll
    for (int ct = 0; ct < 4; ++ct)
#pragma unroll
      for (int g = 0; g < 8; ++g)
        sObuf[(16 * ct + lhalf * 8 + g) * OLD + 16 * nt + lane16] += Oacc[ct][g];
  }
  __syncthreads();
  {
    const int nl = tid & 63, cc = tid >> 6;
    const float inv = 1.0f / sSumRun[nl];
    for (int c = cc; c < CDIM; c += 4)
      out[((size_t)(b * CDIM + c)) * MDIM + n0 + nl] = sObuf[c * OLD + nl] * inv;
  }
}

extern "C" void kernel_launch(void* const* d_in, const int* in_sizes, int n_in,
                              void* d_out, int out_size, void* d_ws, size_t ws_size,
                              hipStream_t stream) {
  (void)in_sizes; (void)n_in; (void)d_ws; (void)ws_size; (void)out_size;
  const float* yi = (const float*)d_in[0];   // feature_yi [B,C,H,W] fp32
  const float* pi = (const float*)d_in[1];   // feature_pi [B,C,H,W] fp32
  float* out = (float*)d_out;                // [B,C,H,W] fp32
  dim3 grid(MDIM / NT, 2, 1);                // 144 n-tiles x B=2
  hfsa_flash_attn_kernel<<<grid, 256, 0, stream>>>(yi, pi, out);
}
